// Llama3Model_1554778161509
// MI455X (gfx1250) — matrix operand order repaired and checked by reading, not compiled, not run on hardware
//
#include <hip/hip_runtime.h>
#include <hip/hip_bf16.h>
#include <math.h>

// ---------------- model constants (match reference) ----------------
#define CV    32000   // vocab
#define CD    1024    // model dim
#define CH    16      // heads
#define CKVH  4       // kv heads
#define CHD   64      // head dim
#define CF    4096    // ffn dim
#define CL    2       // layers
#define CB    2       // batch
#define CS    2048    // seq
#define CT    (CB*CS) // tokens = 4096
#define CREP  (CH/CKVH)
#define CEPS  1e-5f
#define CSCALE (0.125f)              // 1/sqrt(64)
#define LOG_ROPE_BASE 13.1223633774f // ln(500000)

typedef __attribute__((ext_vector_type(16))) __bf16 v16bf;
typedef __attribute__((ext_vector_type(8)))  __bf16 v8bf;
typedef __attribute__((ext_vector_type(8)))  float  v8f;

#define BF16CAT(lo, hi) __builtin_shufflevector((lo), (hi), \
    0,1,2,3,4,5,6,7,8,9,10,11,12,13,14,15)

// ---------------- bf16 helpers (bit-exact RNE) ----------------
__device__ __forceinline__ unsigned short f32_to_bf16_bits(float f) {
    unsigned int u = __builtin_bit_cast(unsigned int, f);
    unsigned int r = u + 0x7FFFu + ((u >> 16) & 1u);
    return (unsigned short)(r >> 16);
}

// ---------------- weight convert + transpose: Wt[N,K](bf16) = W[K,N](f32) ---
__global__ void convT_kernel(const float* __restrict__ W,
                             unsigned short* __restrict__ Wt,
                             int K, int N) {
    size_t gid = (size_t)blockIdx.x * blockDim.x + threadIdx.x;
    if (gid >= (size_t)K * N) return;
    int n = (int)(gid / K);
    int k = (int)(gid - (size_t)n * K);
    Wt[gid] = f32_to_bf16_bits(W[(size_t)k * N + n]);
}

// ---------------- elementwise f32 -> bf16 ----------------
__global__ void cvt_bf16_kernel(const float* __restrict__ src,
                                unsigned short* __restrict__ dst, int n) {
    int gid = blockIdx.x * blockDim.x + threadIdx.x;
    if (gid < n) dst[gid] = f32_to_bf16_bits(src[gid]);
}

// ---------------- V transpose: Vt[b,kvh,hd,S](bf16) = V[b*S+s,kvh,hd](f32) ---
__global__ void transposeV_kernel(const float* __restrict__ V,
                                  unsigned short* __restrict__ Vt) {
    const int gid = blockIdx.x * 256 + threadIdx.x;
    const int s = gid & (CS - 1);
    int r = gid >> 11;                 // CS = 2048
    const int d = r & (CHD - 1);
    r >>= 6;                           // CHD = 64
    const int kvh = r & (CKVH - 1);
    const int b = r >> 2;              // CKVH = 4
    Vt[gid] = f32_to_bf16_bits(V[(((size_t)(b * CS + s)) * CKVH + kvh) * CHD + d]);
}

// ---------------- embedding gather ----------------
__global__ void embed_kernel(const int* __restrict__ idx,
                             const float* __restrict__ emb,
                             float* __restrict__ x) {
    const int t = blockIdx.x;
    const int row = idx[t];
    const float* src = emb + (size_t)row * CD;
    float* dst = x + (size_t)t * CD;
    for (int i = threadIdx.x; i < CD; i += blockDim.x) dst[i] = src[i];
}

// ---------------- RMSNorm -> bf16 activations ----------------
__global__ void rmsnorm_kernel(const float* __restrict__ x,
                               const float* __restrict__ w,
                               unsigned short* __restrict__ out) {
    const int t = blockIdx.x;
    const float* xp = x + (size_t)t * CD;
    __shared__ float red[256];
    float ss = 0.f;
    for (int i = threadIdx.x; i < CD; i += 256) { float v = xp[i]; ss += v * v; }
    red[threadIdx.x] = ss;
    __syncthreads();
    for (int off = 128; off > 0; off >>= 1) {
        if (threadIdx.x < off) red[threadIdx.x] += red[threadIdx.x + off];
        __syncthreads();
    }
    const float inv = rsqrtf(red[0] * (1.0f / CD) + CEPS);
    unsigned short* op = out + (size_t)t * CD;
    for (int i = threadIdx.x; i < CD; i += 256)
        op[i] = f32_to_bf16_bits(xp[i] * inv * w[i]);
}

// ---------------- bf16 WMMA GEMM:  C[M,N] (+)= A[M,K](bf16) * Wt[N,K](bf16)
// block = 256 threads = 8 waves; block tile 16(M) x 128(N); wave tile 16x16; K step 32.
// A tile (16x32 bf16 = 1KB) staged into LDS via CDNA5 async copy by waves 0-1.
template <bool ADD>
__global__ void gemm_bf16_kernel(const unsigned short* __restrict__ A,
                                 const unsigned short* __restrict__ Wt,
                                 float* __restrict__ C,
                                 int M, int N, int K) {
    __shared__ __align__(16) unsigned short tileA[16 * 32];
    const int tid  = threadIdx.x;
    const int lane = tid & 31;
    const int wave = tid >> 5;
    const int n0 = blockIdx.x * 128 + wave * 16;
    const int m0 = blockIdx.y * 16;
    const int mr = lane & 15;
    const int kh = lane >> 4;

    const unsigned int ldsbase = (unsigned int)(uintptr_t)(&tileA[0]);

    v8f acc;
    if (ADD) {
#pragma unroll
        for (int v = 0; v < 8; ++v)
            acc[v] = C[(size_t)(m0 + v + 8 * kh) * N + n0 + mr];
    } else {
#pragma unroll
        for (int v = 0; v < 8; ++v) acc[v] = 0.f;
    }

    for (int k0 = 0; k0 < K; k0 += 32) {
        if (wave < 2) {
            const unsigned int ldsoff = ldsbase + (unsigned int)tid * 16;
            const unsigned short* gp =
                A + (size_t)(m0 + (tid >> 2)) * K + k0 + (tid & 3) * 8;
            asm volatile("global_load_async_to_lds_b128 %0, %1, off"
                         :: "v"(ldsoff), "v"(gp) : "memory");
            asm volatile("s_wait_asynccnt 0x0" ::: "memory");
        }
        __syncthreads();

        const v8bf alo = *(const v8bf*)&tileA[mr * 32 + 16 * kh];
        const v8bf ahi = *(const v8bf*)&tileA[mr * 32 + 16 * kh + 8];
        const v16bf a = BF16CAT(alo, ahi);

        const unsigned short* wtp = Wt + (size_t)(n0 + mr) * K + k0 + 16 * kh;
        const v16bf b = *(const v16bf*)wtp;
        if (k0 + 32 < K) __builtin_prefetch(wtp + 32, 0, 1);

        __syncthreads();

        acc = __builtin_amdgcn_wmma_f32_16x16x32_bf16(
            false, a, false, b, (short)0, acc, false, false);
    }

#pragma unroll
    for (int v = 0; v < 8; ++v)
        C[(size_t)(m0 + v + 8 * kh) * N + n0 + mr] = acc[v];
}

// ---------------- fused gate/up SwiGLU GEMM: H = silu(A*Wg) * (A*Wu), bf16 out
__global__ void gateup_kernel(const unsigned short* __restrict__ A,
                              const unsigned short* __restrict__ WgT,
                              const unsigned short* __restrict__ WuT,
                              unsigned short* __restrict__ Hout,
                              int M, int N, int K) {
    const int lane = threadIdx.x & 31;
    const int wave = threadIdx.x >> 5;
    const int n0 = blockIdx.x * 128 + wave * 16;
    const int m0 = blockIdx.y * 16;
    const int mr = lane & 15;
    const int kh = lane >> 4;

    v8f accg, accu;
#pragma unroll
    for (int v = 0; v < 8; ++v) { accg[v] = 0.f; accu[v] = 0.f; }

    for (int k0 = 0; k0 < K; k0 += 32) {
        const unsigned short* ap = A + (size_t)(m0 + mr) * K + k0 + 16 * kh;
        const v8bf alo = *(const v8bf*)ap;
        const v8bf ahi = *(const v8bf*)(ap + 8);
        const v16bf a = BF16CAT(alo, ahi);

        const unsigned short* gp = WgT + (size_t)(n0 + mr) * K + k0 + 16 * kh;
        const unsigned short* up = WuT + (size_t)(n0 + mr) * K + k0 + 16 * kh;
        const v16bf bg = *(const v16bf*)gp;
        const v16bf bu = *(const v16bf*)up;
        if (k0 + 32 < K) {
            __builtin_prefetch(gp + 32, 0, 1);
            __builtin_prefetch(up + 32, 0, 1);
        }
        accg = __builtin_amdgcn_wmma_f32_16x16x32_bf16(
            false, a, false, bg, (short)0, accg, false, false);
        accu = __builtin_amdgcn_wmma_f32_16x16x32_bf16(
            false, a, false, bu, (short)0, accu, false, false);
    }

#pragma unroll
    for (int v = 0; v < 8; ++v) {
        float g = accg[v];
        float s = g * __frcp_rn(1.0f + __expf(-g)); // silu
        Hout[(size_t)(m0 + v + 8 * kh) * N + n0 + mr] = f32_to_bf16_bits(s * accu[v]);
    }
}

// ---------------- RoPE in place ----------------
__global__ void rope_kernel(float* __restrict__ x, int heads) {
    const int gid = blockIdx.x * blockDim.x + threadIdx.x;
    const int d  = gid & 31;
    const int hh = (gid >> 5) % heads;
    const int t  = gid / (heads * 32);
    const int pos = t % CS;
    const float inv_freq = __expf(-((float)d * (1.0f / 32.0f)) * LOG_ROPE_BASE);
    const float fr = (float)pos * inv_freq;
    float c, s;
    __sincosf(fr, &s, &c);
    float* p = x + ((size_t)t * heads + hh) * CHD;
    const float x0 = p[d];
    const float x1 = p[d + 32];
    p[d]      = x0 * c - x1 * s;
    p[d + 32] = x1 * c + x0 * s;
}

// ---------------- WMMA flash attention (causal GQA, online softmax) --------
// One wave per 16-query tile; 32 keys per iteration.
// Qb [B*S, CH, 64] bf16 ; Kb [B*S, CKVH, 64] bf16 ; Vt [B,CKVH,64,S] bf16
// ctx [B*S, CH*64] bf16
__global__ void __launch_bounds__(256)
flash_attn_kernel(const unsigned short* __restrict__ Qb,
                  const unsigned short* __restrict__ Kb,
                  const unsigned short* __restrict__ Vt,
                  unsigned short* __restrict__ ctx) {
    __shared__ __align__(16) unsigned short pbuf[8 * 16 * 32];
    const int lane = threadIdx.x & 31;
    const int wave = threadIdx.x >> 5;
    const int wgid = blockIdx.x * 8 + wave;
    const int qt = wgid & (CS / 16 - 1);   // 128 q-tiles
    const int bh = wgid >> 7;
    const int b  = bh / CH;
    const int h  = bh % CH;
    const int kvh = h / CREP;
    const int mr = lane & 15;
    const int kh = lane >> 4;

    unsigned short* pw = pbuf + wave * (16 * 32);

    // Q fragments (hd 0..31 and 32..63), reused over all key blocks
    const unsigned short* qp =
        Qb + ((size_t)(b * CS + qt * 16 + mr) * CH + h) * CHD;
    const v16bf a0 = BF16CAT(*(const v8bf*)(qp +       16 * kh),
                             *(const v8bf*)(qp + 8 +  16 * kh));
    const v16bf a1 = BF16CAT(*(const v8bf*)(qp + 32 +  16 * kh),
                             *(const v8bf*)(qp + 40 +  16 * kh));

    v8f acc[4];
#pragma unroll
    for (int n = 0; n < 4; ++n)
#pragma unroll
        for (int v = 0; v < 8; ++v) acc[n][v] = 0.f;

    float mrow[8], lrow[8];
#pragma unroll
    for (int v = 0; v < 8; ++v) { mrow[v] = -3.4e38f; lrow[v] = 0.f; }

    for (int j0 = 0; j0 <= qt * 16; j0 += 32) {
        // ---- scores: two 16-key tiles, 2 WMMAs each (hd halves) ----
        v8f s[2];
#pragma unroll
        for (int t = 0; t < 2; ++t) {
            const int key = j0 + 16 * t + mr;
            const int keyc = key < CS ? key : CS - 1;   // clamp masked-out loads
            const unsigned short* kp =
                Kb + ((size_t)(b * CS + keyc) * CKVH + kvh) * CHD + 16 * kh;
            const v16bf b0 = *(const v16bf*)kp;
            const v16bf b1 = *(const v16bf*)(kp + 32);
            v8f z;
#pragma unroll
            for (int v = 0; v < 8; ++v) z[v] = 0.f;
            z = __builtin_amdgcn_wmma_f32_16x16x32_bf16(
                false, a0, false, b0, (short)0, z, false, false);
            z = __builtin_amdgcn_wmma_f32_16x16x32_bf16(
                false, a1, false, b1, (short)0, z, false, false);
            s[t] = z;
        }

        // ---- scale, causal mask, row max (shfl across 16-lane half) ----
        float mnew[8];
#pragma unroll
        for (int v = 0; v < 8; ++v) {
            const int qrow = qt * 16 + v + 8 * kh;
            float s0 = (j0 + mr      <= qrow) ? s[0][v] * CSCALE : -3.0e38f;
            float s1 = (j0 + 16 + mr <= qrow) ? s[1][v] * CSCALE : -3.0e38f;
            s[0][v] = s0; s[1][v] = s1;
            float mx = fmaxf(s0, s1);
            mx = fmaxf(mx, __shfl_xor(mx, 1, 32));
            mx = fmaxf(mx, __shfl_xor(mx, 2, 32));
            mx = fmaxf(mx, __shfl_xor(mx, 4, 32));
            mx = fmaxf(mx, __shfl_xor(mx, 8, 32));
            mnew[v] = fmaxf(mrow[v], mx);
        }

        // ---- p = exp(s-m), row sum, rescale acc, stage P to LDS ----
#pragma unroll
        for (int v = 0; v < 8; ++v) {
            const float scale = __expf(mrow[v] - mnew[v]);
            const float p0 = __expf(s[0][v] - mnew[v]);
            const float p1 = __expf(s[1][v] - mnew[v]);
            float rs = p0 + p1;
            rs += __shfl_xor(rs, 1, 32);
            rs += __shfl_xor(rs, 2, 32);
            rs += __shfl_xor(rs, 4, 32);
            rs += __shfl_xor(rs, 8, 32);
            lrow[v] = lrow[v] * scale + rs;
            mrow[v] = mnew[v];
#pragma unroll
            for (int n = 0; n < 4; ++n) acc[n][v] *= scale;
            pw[(v + 8 * kh) * 32 + mr]      = f32_to_bf16_bits(p0);
            pw[(v + 8 * kh) * 32 + 16 + mr] = f32_to_bf16_bits(p1);
        }

        // ---- P as A-fragment (LDS is in-order within a wave) ----
        const unsigned short* pr = pw + mr * 32 + 16 * kh;
        const v16bf aP = BF16CAT(*(const v8bf*)pr, *(const v8bf*)(pr + 8));

        // ---- ctx += P(16x32) * V(32x64): 4 N-tiles ----
#pragma unroll
        for (int n = 0; n < 4; ++n) {
            const unsigned short* vp =
                Vt + ((size_t)((b * CKVH + kvh) * CHD + n * 16 + mr)) * CS
                   + j0 + 16 * kh;
            const v16bf bv = *(const v16bf*)vp;
            acc[n] = __builtin_amdgcn_wmma_f32_16x16x32_bf16(
                false, aP, false, bv, (short)0, acc[n], false, false);
        }
    }

    // ---- normalize and write ctx (bf16, [tok, h, hd]) ----
#pragma unroll
    for (int v = 0; v < 8; ++v) {
        const float inv = __frcp_rn(lrow[v]);
        const int tok = b * CS + qt * 16 + v + 8 * kh;
        unsigned short* op = ctx + ((size_t)tok * CH + h) * CHD;
#pragma unroll
        for (int n = 0; n < 4; ++n)
            op[n * 16 + mr] = f32_to_bf16_bits(acc[n][v] * inv);
    }
}

// ---------------- host orchestration ----------------
extern "C" void kernel_launch(void* const* d_in, const int* in_sizes, int n_in,
                              void* d_out, int out_size, void* d_ws, size_t ws_size,
                              hipStream_t stream) {
    (void)in_sizes; (void)n_in; (void)out_size; (void)ws_size;
    const int*   in_idx     = (const int*)  d_in[0];
    const float* token_emb  = (const float*)d_in[1];
    const float* wq         = (const float*)d_in[2];
    const float* wk         = (const float*)d_in[3];
    const float* wv         = (const float*)d_in[4];
    const float* wo         = (const float*)d_in[5];
    const float* w_gate     = (const float*)d_in[6];
    const float* w_up       = (const float*)d_in[7];
    const float* w_down     = (const float*)d_in[8];
    const float* norm_attn  = (const float*)d_in[9];
    const float* norm_ff    = (const float*)d_in[10];
    const float* norm_final = (const float*)d_in[11];
    const float* w_out      = (const float*)d_in[12];
    float* logits = (float*)d_out;

    // ---- workspace carve-up ----
    char* ws = (char*)d_ws;
    size_t off = 0;
    float* x = (float*)(ws + off);                      off += (size_t)CT * CD * 4;
    unsigned short* nbuf = (unsigned short*)(ws + off); off += (size_t)CT * CD * 2;
    float* qb = (float*)(ws + off);                     off += (size_t)CT * CH   * CHD * 4;
    float* kb = (float*)(ws + off);                     off += (size_t)CT * CKVH * CHD * 4;
    float* vb = (float*)(ws + off);                     off += (size_t)CT * CKVH * CHD * 4;
    unsigned short* ctx = (unsigned short*)(ws + off);  off += (size_t)CT * CD * 2;
    unsigned short* hb  = (unsigned short*)(ws + off);  off += (size_t)CT * CF * 2;
    unsigned short* qbf = (unsigned short*)(ws + off);  off += (size_t)CT * CH   * CHD * 2;
    unsigned short* kbf = (unsigned short*)(ws + off);  off += (size_t)CT * CKVH * CHD * 2;
    unsigned short* vtb = (unsigned short*)(ws + off);  off += ((size_t)CT * CKVH * CHD + 64) * 2;

    unsigned short* wqT   = (unsigned short*)(ws + off); off += (size_t)CL * CD * CD * 2;
    unsigned short* wkT   = (unsigned short*)(ws + off); off += (size_t)CL * CD * CKVH * CHD * 2;
    unsigned short* wvT   = (unsigned short*)(ws + off); off += (size_t)CL * CD * CKVH * CHD * 2;
    unsigned short* woT   = (unsigned short*)(ws + off); off += (size_t)CL * CD * CD * 2;
    unsigned short* wgT   = (unsigned short*)(ws + off); off += (size_t)CL * CD * CF * 2;
    unsigned short* wuT   = (unsigned short*)(ws + off); off += (size_t)CL * CD * CF * 2;
    unsigned short* wdT   = (unsigned short*)(ws + off); off += (size_t)CL * CF * CD * 2;
    unsigned short* woutT = (unsigned short*)(ws + off); off += (size_t)CV * CD * 2;

    const dim3 blk(256);
    auto cvt = [&](const float* src, unsigned short* dst, int Kd, int Nd) {
        size_t total = (size_t)Kd * Nd;
        convT_kernel<<<(unsigned)((total + 255) / 256), blk, 0, stream>>>(src, dst, Kd, Nd);
    };

    for (int l = 0; l < CL; ++l) {
        cvt(wq     + (size_t)l * CD * CD,          wqT + (size_t)l * CD * CD,          CD, CD);
        cvt(wk     + (size_t)l * CD * CKVH * CHD,  wkT + (size_t)l * CD * CKVH * CHD,  CD, CKVH * CHD);
        cvt(wv     + (size_t)l * CD * CKVH * CHD,  wvT + (size_t)l * CD * CKVH * CHD,  CD, CKVH * CHD);
        cvt(wo     + (size_t)l * CD * CD,          woT + (size_t)l * CD * CD,          CD, CD);
        cvt(w_gate + (size_t)l * CD * CF,          wgT + (size_t)l * CD * CF,          CD, CF);
        cvt(w_up   + (size_t)l * CD * CF,          wuT + (size_t)l * CD * CF,          CD, CF);
        cvt(w_down + (size_t)l * CF * CD,          wdT + (size_t)l * CF * CD,          CF, CD);
    }
    cvt(w_out, woutT, CD, CV);

    const dim3 gD   (CD / 128,        CT / 16);
    const dim3 gKV  ((CKVH*CHD)/128,  CT / 16);
    const dim3 gF   (CF / 128,        CT / 16);
    const dim3 gV   (CV / 128,        CT / 16);

    embed_kernel<<<CT, blk, 0, stream>>>(in_idx, token_emb, x);

    const int nQ  = CT * CH   * CHD;   // 4M
    const int nKV = CT * CKVH * CHD;   // 1M

    for (int l = 0; l < CL; ++l) {
        const unsigned short* lwq = wqT + (size_t)l * CD * CD;
        const unsigned short* lwk = wkT + (size_t)l * CD * CKVH * CHD;
        const unsigned short* lwv = wvT + (size_t)l * CD * CKVH * CHD;
        const unsigned short* lwo = woT + (size_t)l * CD * CD;
        const unsigned short* lg  = wgT + (size_t)l * CD * CF;
        const unsigned short* lu  = wuT + (size_t)l * CD * CF;
        const unsigned short* ld  = wdT + (size_t)l * CF * CD;

        // attention sub-block
        rmsnorm_kernel<<<CT, blk, 0, stream>>>(x, norm_attn + (size_t)l * CD, nbuf);
        gemm_bf16_kernel<false><<<gD,  blk, 0, stream>>>(nbuf, lwq, qb, CT, CD,         CD);
        gemm_bf16_kernel<false><<<gKV, blk, 0, stream>>>(nbuf, lwk, kb, CT, CKVH * CHD, CD);
        gemm_bf16_kernel<false><<<gKV, blk, 0, stream>>>(nbuf, lwv, vb, CT, CKVH * CHD, CD);
        rope_kernel<<<(CT * CH   * 32) / 256, blk, 0, stream>>>(qb, CH);
        rope_kernel<<<(CT * CKVH * 32) / 256, blk, 0, stream>>>(kb, CKVH);
        cvt_bf16_kernel<<<nQ  / 256, blk, 0, stream>>>(qb, qbf, nQ);
        cvt_bf16_kernel<<<nKV / 256, blk, 0, stream>>>(kb, kbf, nKV);
        transposeV_kernel<<<nKV / 256, blk, 0, stream>>>(vb, vtb);
        flash_attn_kernel<<<(CB * CH * (CS / 16)) / 8, blk, 0, stream>>>(qbf, kbf, vtb, ctx);
        gemm_bf16_kernel<true><<<gD, blk, 0, stream>>>(ctx, lwo, x, CT, CD, CD); // x += ctx@wo

        // ffn sub-block
        rmsnorm_kernel<<<CT, blk, 0, stream>>>(x, norm_ff + (size_t)l * CD, nbuf);
        gateup_kernel<<<gF, blk, 0, stream>>>(nbuf, lg, lu, hb, CT, CF, CD);
        gemm_bf16_kernel<true><<<gD, blk, 0, stream>>>(hb, ld, x, CT, CD, CF); // x += h@w_down
    }

    rmsnorm_kernel<<<CT, blk, 0, stream>>>(x, norm_final, nbuf);
    gemm_bf16_kernel<false><<<gV, blk, 0, stream>>>(nbuf, woutT, logits, CT, CV, CD);
}
